// ABHUE_66185446032058
// MI455X (gfx1250) — compile-verified, matrix operand order checked by reading, not run on hardware
//
#include <hip/hip_runtime.h>
#include <hip/hip_bf16.h>
#include <math.h>

// ---------------------------------------------------------------------------
// ABHUE hierarchical LSTM for MI455X (gfx1250, wave32, WMMA)
//   S=63 sentences, W=128 words, E=200, H=200, 4H=800
// Plan:
//   K0 pack : fp32 -> f16 padded weight/activation layouts in d_ws
//   K1 gemm : Xg[8192][800] = x @ Wih^T + b  (ctx rows 0..8063, main rows 8064..8191)
//             full-chip v_wmma_f32_16x16x32_f16
//   K2 lstm : persistent single-workgroup (1024 thr = 32 waves) kernel:
//             128-step batched word recurrence (state in 295KB LDS, Whh in L2),
//             then 32-step prev/post sentence LSTMs, then final FC -> d_out
// ---------------------------------------------------------------------------

typedef _Float16 half_t;
typedef __attribute__((ext_vector_type(8)))  _Float16 v8h;
typedef __attribute__((ext_vector_type(16))) _Float16 v16h;
typedef __attribute__((ext_vector_type(8)))  float    v8f;

// ---- problem constants ----
#define KP   224            // K pad of 200 (7 chunks of 32)
#define KS   448            // sentence concat K pad (2*224, 14 chunks)
#define G4   800            // 4*H
#define MR   8192           // 63*128 ctx rows + 128 main rows
#define MID  31

// ---- workspace layout (bytes, all 256-aligned) ----
#define OFF_XG     ((size_t)0)                                  // fp32 [8192][800]
#define OFF_XF16   (OFF_XG    + (size_t)MR*G4*4)                // f16  [8192][224]
#define OFF_WHH    (OFF_XF16  + (size_t)MR*KP*2)                // f16  [2][800][224] ctx,main
#define OFF_WIH    (OFF_WHH   + (size_t)2*G4*KP*2)              // f16  [2][800][224] ctx,main
#define OFF_WSENT  (OFF_WIH   + (size_t)2*G4*KP*2)              // f16  [2][800][448] prev,post
#define OFF_BIAS   (OFF_WSENT + (size_t)2*G4*KS*2)              // fp32 [2][800] ctx,main

// ---- LDS layout for K2 (dynamic shared) ----
#define L_HBUF     0                        // f16  [80][224]   35840 B
#define L_GATES    35840                    // fp32 [64][800]  204800 B
#define L_MGATES   (35840+204800)           // fp32 [800]        3200 B
#define L_CBUF     (35840+204800+3200)      // fp32 [64][200]   51200 B
#define SMEM_BYTES (35840+204800+3200+51200)   // 295040 <= 320KB/WGP
// sentence-phase reuse of the GATES region:
#define L_ABUF     L_GATES                  // f16  [2][16][448] 28672 B
#define L_GSENT    (L_GATES+28672)          // fp32 [2][800]      6400 B
#define L_CSENT    L_MGATES                 // fp32 [2][200]

// load 16 halves for a WMMA operand lane: two 16B chunks at p and p+16
__device__ __forceinline__ v16h ld16(const half_t* p) {
    struct alignas(32) Pair { v8h lo, hi; } t;
    t.lo = *reinterpret_cast<const v8h*>(p);
    t.hi = *reinterpret_cast<const v8h*>(p + 16);
    return __builtin_bit_cast(v16h, t);
}

__device__ __forceinline__ v8f wmma16(v16h a, v16h b, v8f c) {
    return __builtin_amdgcn_wmma_f32_16x16x32_f16(false, a, false, b, (short)0, c,
                                                  false, false);
}

__device__ __forceinline__ float sigm(float v) { return 1.0f / (1.0f + expf(-v)); }

// ---------------------------------------------------------------------------
// K0: pack fp32 inputs into padded f16 layouts + fused biases
// ---------------------------------------------------------------------------
__global__ void abhue_pack(
    const float* __restrict__ x,
    const float* __restrict__ Wih_ctx,  const float* __restrict__ Whh_ctx,
    const float* __restrict__ bih_ctx,  const float* __restrict__ bhh_ctx,
    const float* __restrict__ Wih_main, const float* __restrict__ Whh_main,
    const float* __restrict__ bih_main, const float* __restrict__ bhh_main,
    const float* __restrict__ Wih_prev, const float* __restrict__ Whh_prev,
    const float* __restrict__ Wih_post, const float* __restrict__ Whh_post,
    char* __restrict__ ws)
{
    half_t* XF16  = (half_t*)(ws + OFF_XF16);
    half_t* WHH   = (half_t*)(ws + OFF_WHH);
    half_t* WIH   = (half_t*)(ws + OFF_WIH);
    half_t* WSENT = (half_t*)(ws + OFF_WSENT);
    float*  BIAS  = (float*) (ws + OFF_BIAS);

    const size_t N0 = (size_t)MR * KP;          // 1,835,008
    const size_t N1 = (size_t)2 * G4 * KP;      //   358,400 (Whh)
    const size_t N2 = N1;                       //   358,400 (Wih)
    const size_t N3 = (size_t)2 * G4 * KS;      //   716,800 (sentence concat W)
    const size_t N4 = (size_t)2 * G4;           //     1,600 (bias)
    const size_t total = N0 + N1 + N2 + N3 + N4;

    for (size_t i = (size_t)blockIdx.x * blockDim.x + threadIdx.x; i < total;
         i += (size_t)gridDim.x * blockDim.x) {
        if (i < N0) {                                   // XF16
            size_t r = i / KP, k = i % KP;
            int s, t;
            if (r < 8064) { s = (int)(r / 128); t = (int)(r % 128); }
            else          { s = MID;            t = (int)(r - 8064); }
            float v = (k < 200) ? x[((size_t)s * 128 + t) * 200 + k] : 0.0f;
            XF16[i] = (half_t)v;
        } else if (i < N0 + N1) {                       // WHH (ctx, main)
            size_t j = i - N0;
            int sel = (int)(j / ((size_t)G4 * KP));
            size_t rem = j % ((size_t)G4 * KP);
            int row = (int)(rem / KP), k = (int)(rem % KP);
            const float* src = sel ? Whh_main : Whh_ctx;
            WHH[j] = (half_t)((k < 200) ? src[row * 200 + k] : 0.0f);
        } else if (i < N0 + N1 + N2) {                  // WIH (ctx, main)
            size_t j = i - N0 - N1;
            int sel = (int)(j / ((size_t)G4 * KP));
            size_t rem = j % ((size_t)G4 * KP);
            int row = (int)(rem / KP), k = (int)(rem % KP);
            const float* src = sel ? Wih_main : Wih_ctx;
            WIH[j] = (half_t)((k < 200) ? src[row * 200 + k] : 0.0f);
        } else if (i < N0 + N1 + N2 + N3) {             // WSENT = [Whh | Wih] (prev,post)
            size_t j = i - N0 - N1 - N2;
            int sel = (int)(j / ((size_t)G4 * KS));
            size_t rem = j % ((size_t)G4 * KS);
            int row = (int)(rem / KS), k = (int)(rem % KS);
            const float* Wh = sel ? Whh_post : Whh_prev;
            const float* Wi = sel ? Wih_post : Wih_prev;
            float v = 0.0f;
            if (k < 200)                 v = Wh[row * 200 + k];
            else if (k >= 224 && k < 424) v = Wi[row * 200 + (k - 224)];
            WSENT[j] = (half_t)v;
        } else {                                        // fused biases
            size_t j = i - N0 - N1 - N2 - N3;
            int sel = (int)(j / G4);
            int col = (int)(j % G4);
            BIAS[j] = sel ? (bih_main[col] + bhh_main[col])
                          : (bih_ctx[col]  + bhh_ctx[col]);
        }
    }
}

// ---------------------------------------------------------------------------
// K1: Xg = XF16 @ Wih^T + bias   (M=8192, N=800, K=224)  -- full-chip WMMA
// one wave per 16x16 output tile; 3200 blocks * 8 waves = 25600 tiles
// ---------------------------------------------------------------------------
__global__ __launch_bounds__(256) void abhue_gemm1(char* __restrict__ ws)
{
    const half_t* XF16 = (const half_t*)(ws + OFF_XF16);
    const half_t* WIH  = (const half_t*)(ws + OFF_WIH);
    const float*  BIAS = (const float*) (ws + OFF_BIAS);
    float*        XG   = (float*)      (ws + OFF_XG);

    int wave = blockIdx.x * 8 + (threadIdx.x >> 5);
    int lane = threadIdx.x & 31;
    int mt = wave / 50;
    int nt = wave - mt * 50;
    int sel = (mt >= 504) ? 1 : 0;            // tiles 504..511 are the main-LSTM rows
    int l15 = lane & 15;
    int kh  = (lane >> 4) * 8;                // A/B k-half split per ISA layout

    const half_t* Ap = XF16 + (size_t)(mt * 16 + l15) * KP + kh;
    const half_t* Bp = WIH  + (size_t)sel * G4 * KP + (size_t)(nt * 16 + l15) * KP + kh;

    v8f acc = {};
#pragma unroll
    for (int kc = 0; kc < 7; ++kc)
        acc = wmma16(ld16(Ap + kc * 32), ld16(Bp + kc * 32), acc);

    float bias = BIAS[sel * G4 + nt * 16 + l15];
    int m0 = mt * 16 + (lane >> 4) * 8;       // C layout: m = r + 8*lane[4]
    int n  = nt * 16 + l15;
#pragma unroll
    for (int r = 0; r < 8; ++r)
        XG[(size_t)(m0 + r) * G4 + n] = acc[r] + bias;
}

// ---------------------------------------------------------------------------
// K2: persistent recurrence kernel (single workgroup, 32 waves)
// ---------------------------------------------------------------------------
__global__ __launch_bounds__(1024) void abhue_lstm(
    const float* __restrict__ h0s,      const float* __restrict__ c0s,
    const float* __restrict__ h_prev0,  const float* __restrict__ c_prev0,
    const float* __restrict__ h_post0,  const float* __restrict__ c_post0,
    const float* __restrict__ bih_prev, const float* __restrict__ bhh_prev,
    const float* __restrict__ bih_post, const float* __restrict__ bhh_post,
    const float* __restrict__ Wfc,      const float* __restrict__ bfc,
    char* __restrict__ ws, float* __restrict__ out)
{
    extern __shared__ char smem[];
    half_t* hbuf   = (half_t*)(smem + L_HBUF);    // [80][224]; rows 0..62 ctx, 64 main
    float*  gates  = (float*) (smem + L_GATES);   // [64][800]
    float*  mgates = (float*) (smem + L_MGATES);  // [800] main-row gates
    float*  cbuf   = (float*) (smem + L_CBUF);    // [64][200]; row 63 = main c
    half_t* abuf   = (half_t*)(smem + L_ABUF);    // [2][16][448] sentence A tiles
    float*  gsent  = (float*) (smem + L_GSENT);   // [2][800]
    float*  csent  = (float*) (smem + L_CSENT);   // [2][200]

    const half_t* WHH   = (const half_t*)(ws + OFF_WHH);
    const half_t* WSENT = (const half_t*)(ws + OFF_WSENT);
    const float*  XG    = (const float*) (ws + OFF_XG);

    const int tid  = threadIdx.x;
    const int wave = tid >> 5;
    const int lane = tid & 31;
    const int l15  = lane & 15;
    const int kh   = (lane >> 4) * 8;

    // ---- init LDS state ----
    for (int i = tid; i < 80 * KP; i += 1024) {
        int row = i / KP, k = i % KP;
        float v = 0.0f;
        if (k < 200) {
            if (row < 63)       v = h0s[row * 200 + k];
            else if (row == 64) v = h0s[MID * 200 + k];
        }
        hbuf[i] = (half_t)v;
    }
    for (int i = tid; i < 64 * 200; i += 1024) {
        int row = i / 200, j = i % 200;
        cbuf[i] = (row < 63) ? c0s[row * 200 + j] : c0s[MID * 200 + j];
    }
    __syncthreads();

    // ================= word-level recurrence: 128 steps =================
    for (int t = 0; t < 128; ++t) {
        // GEMM: gates = h @ Whh^T ; 5 M-tiles x 50 N-tiles
        for (int job = wave; job < 250; job += 32) {
            int mt = job / 50, nt = job - mt * 50;
            int sel  = (mt == 4) ? 1 : 0;                 // tile 4 = main weights
            int arow = ((mt == 4) ? 64 : mt * 16) + l15;
            const half_t* Ap = hbuf + arow * KP + kh;
            const half_t* Bp = WHH + (size_t)sel * G4 * KP
                                   + (size_t)(nt * 16 + l15) * KP + kh;
            v8f acc = {};
#pragma unroll
            for (int kc = 0; kc < 7; ++kc)
                acc = wmma16(ld16(Ap + kc * 32), ld16(Bp + kc * 32), acc);
            if (mt < 4) {
                int m0 = mt * 16 + (lane >> 4) * 8;
                int n  = nt * 16 + l15;
#pragma unroll
                for (int r = 0; r < 8; ++r)
                    gates[(m0 + r) * G4 + n] = acc[r];
            } else if (lane < 16) {                        // only row 0 is live
                mgates[nt * 16 + lane] = acc[0];
            }
        }
        __syncthreads();

        // elementwise: 63*200 ctx units + 200 main units
        for (int u = tid; u < 12800; u += 1024) {
            float gi, gf, gg, go, c;
            if (u < 12600) {
                int row = u / 200, j = u - row * 200;
                const float* xg = XG + ((size_t)(row * 128 + t)) * G4;
                gi = gates[row * G4 + j]       + xg[j];
                gf = gates[row * G4 + 200 + j] + xg[200 + j];
                gg = gates[row * G4 + 400 + j] + xg[400 + j];
                go = gates[row * G4 + 600 + j] + xg[600 + j];
                c  = cbuf[row * 200 + j];
                float fi = sigm(gi), ff = sigm(gf), fg = tanhf(gg), fo = sigm(go);
                float cn = ff * c + fi * fg;
                cbuf[row * 200 + j] = cn;
                hbuf[row * KP + j]  = (half_t)(fo * tanhf(cn));
            } else {
                int j = u - 12600;
                const float* xg = XG + ((size_t)(8064 + t)) * G4;
                gi = mgates[j]       + xg[j];
                gf = mgates[200 + j] + xg[200 + j];
                gg = mgates[400 + j] + xg[400 + j];
                go = mgates[600 + j] + xg[600 + j];
                c  = cbuf[63 * 200 + j];
                float fi = sigm(gi), ff = sigm(gf), fg = tanhf(gg), fo = sigm(go);
                float cn = ff * c + fi * fg;
                cbuf[63 * 200 + j] = cn;
                hbuf[64 * KP + j]  = (half_t)(fo * tanhf(cn));
            }
        }
        __syncthreads();
    }

    // sent_emb[MID] = main LSTM output
    for (int i = tid; i < KP; i += 1024) hbuf[MID * KP + i] = hbuf[64 * KP + i];
    // zero sentence A-tiles (rows 1..15 and pads must be 0)
    for (int i = tid; i < 2 * 16 * KS; i += 1024) abuf[i] = (half_t)0;
    __syncthreads();
    for (int i = tid; i < 200; i += 1024) {
        abuf[0 * 16 * KS + i] = (half_t)h_prev0[i];
        abuf[1 * 16 * KS + i] = (half_t)h_post0[i];
        csent[i]       = c_prev0[i];
        csent[200 + i] = c_post0[i];
    }
    __syncthreads();

    // ================= sentence-level: 32 steps, prev & post =================
    for (int u = 0; u < 32; ++u) {
        // x part of concat A rows: prev <- sent_emb[u], post <- sent_emb[62-u]
        for (int i = tid; i < KP; i += 1024) {
            abuf[0 * 16 * KS + 224 + i] = hbuf[u * KP + i];
            abuf[1 * 16 * KS + 224 + i] = hbuf[(62 - u) * KP + i];
        }
        __syncthreads();

        // GEMM: [h|x] @ [Whh|Wih]^T ; 2 M-tiles x 50 N-tiles, K=448
        for (int job = wave; job < 100; job += 32) {
            int mt = job / 50, nt = job - mt * 50;
            const half_t* Ap = abuf + mt * 16 * KS + l15 * KS + kh;
            const half_t* Bp = WSENT + (size_t)mt * G4 * KS
                                     + (size_t)(nt * 16 + l15) * KS + kh;
            v8f acc = {};
#pragma unroll
            for (int kc = 0; kc < 14; ++kc)
                acc = wmma16(ld16(Ap + kc * 32), ld16(Bp + kc * 32), acc);
            if (lane < 16) gsent[mt * G4 + nt * 16 + lane] = acc[0];
        }
        __syncthreads();

        for (int v = tid; v < 400; v += 1024) {
            int br = v / 200, j = v - br * 200;
            const float* bi = br ? bih_post : bih_prev;
            const float* bh = br ? bhh_post : bhh_prev;
            float gi = gsent[br * G4 + j]       + bi[j]       + bh[j];
            float gf = gsent[br * G4 + 200 + j] + bi[200 + j] + bh[200 + j];
            float gg = gsent[br * G4 + 400 + j] + bi[400 + j] + bh[400 + j];
            float go = gsent[br * G4 + 600 + j] + bi[600 + j] + bh[600 + j];
            float c  = csent[br * 200 + j];
            float fi = sigm(gi), ff = sigm(gf), fg = tanhf(gg), fo = sigm(go);
            float cn = ff * c + fi * fg;
            csent[br * 200 + j]   = cn;
            abuf[br * 16 * KS + j] = (half_t)(fo * tanhf(cn));
        }
        __syncthreads();
    }

    // ================= final FC: out = [h_prev ; h_post] @ Wfc^T + bfc ========
    for (int e = tid; e < 200; e += 1024) {
        float acc = bfc[e];
        for (int k = 0; k < 200; ++k)
            acc += (float)abuf[k] * Wfc[e * 400 + k];
        for (int k = 0; k < 200; ++k)
            acc += (float)abuf[16 * KS + k] * Wfc[e * 400 + 200 + k];
        out[e] = acc;
    }
}

// ---------------------------------------------------------------------------
extern "C" void kernel_launch(void* const* d_in, const int* in_sizes, int n_in,
                              void* d_out, int out_size, void* d_ws, size_t ws_size,
                              hipStream_t stream) {
    (void)in_sizes; (void)n_in; (void)out_size; (void)ws_size;
    const float* x        = (const float*)d_in[0];
    const float* h0s      = (const float*)d_in[1];
    const float* c0s      = (const float*)d_in[2];
    const float* h_prev0  = (const float*)d_in[3];
    const float* c_prev0  = (const float*)d_in[4];
    const float* h_post0  = (const float*)d_in[5];
    const float* c_post0  = (const float*)d_in[6];
    const float* Wih_ctx  = (const float*)d_in[7];
    const float* Whh_ctx  = (const float*)d_in[8];
    const float* bih_ctx  = (const float*)d_in[9];
    const float* bhh_ctx  = (const float*)d_in[10];
    const float* Wih_main = (const float*)d_in[11];
    const float* Whh_main = (const float*)d_in[12];
    const float* bih_main = (const float*)d_in[13];
    const float* bhh_main = (const float*)d_in[14];
    const float* Wih_prev = (const float*)d_in[15];
    const float* Whh_prev = (const float*)d_in[16];
    const float* bih_prev = (const float*)d_in[17];
    const float* bhh_prev = (const float*)d_in[18];
    const float* Wih_post = (const float*)d_in[19];
    const float* Whh_post = (const float*)d_in[20];
    const float* bih_post = (const float*)d_in[21];
    const float* bhh_post = (const float*)d_in[22];
    const float* Wfc      = (const float*)d_in[23];
    const float* bfc      = (const float*)d_in[24];
    char* ws   = (char*)d_ws;
    float* out = (float*)d_out;

    abhue_pack<<<2048, 256, 0, stream>>>(x,
        Wih_ctx, Whh_ctx, bih_ctx, bhh_ctx,
        Wih_main, Whh_main, bih_main, bhh_main,
        Wih_prev, Whh_prev, Wih_post, Whh_post, ws);

    abhue_gemm1<<<3200, 256, 0, stream>>>(ws);   // 25600 waves = 25600 tiles

    (void)hipFuncSetAttribute(reinterpret_cast<const void*>(abhue_lstm),
                              hipFuncAttributeMaxDynamicSharedMemorySize,
                              (int)SMEM_BYTES);
    abhue_lstm<<<1, 1024, SMEM_BYTES, stream>>>(
        h0s, c0s, h_prev0, c_prev0, h_post0, c_post0,
        bih_prev, bhh_prev, bih_post, bhh_post, Wfc, bfc, ws, out);
}
